// LinODEnet_53334903882245
// MI455X (gfx1250) — compile-verified, matrix operand order checked by reading, not run on hardware
//
#include <hip/hip_runtime.h>
#include <hip/hip_bf16.h>

// ---------------------------------------------------------------------------
// LinODEnet on MI455X (gfx1250, wave32, WMMA)
//
//  kernel 1: pack    -- convert all weight matrices to f16 in workspace
//  kernel 2: expm    -- per-timestep matrix exponential of A*dt (128 blocks x
//                       16 waves, Horner-12 + 6 squarings) in FULL FP32 via
//                       v_wmma_f32_16x16x4_f32, all operands in LDS
//  kernel 3: scan    -- persistent sequential scan; 32 blocks each own a
//                       16-row batch slice and loop over all 128 timesteps,
//                       batch matmuls via v_wmma_f32_16x16x32_f16
// ---------------------------------------------------------------------------

#define BATCH 512
#define LEN   128
#define DOBS  96
#define NLAT  128
#define NB    5

typedef __attribute__((ext_vector_type(16))) _Float16 v16h;
typedef __attribute__((ext_vector_type(8)))  float    v8f;
typedef __attribute__((ext_vector_type(2)))  float    v2f;

// ---- workspace layout (units: _Float16 elements) --------------------------
#define EXP_OFF   0
#define EXP_N     (LEN * NLAT * NLAT)              // 2,097,152  (4 MB)
#define WSEG      (NB * NLAT * NLAT)               // 81,920 per resnet weight
#define ENC1_OFF  (EXP_OFF + EXP_N)
#define ENC2_OFF  (ENC1_OFF + WSEG)
#define DEC1_OFF  (ENC2_OFF + WSEG)
#define DEC2_OFF  (DEC1_OFF + WSEG)
#define WEMB_OFF  (DEC2_OFF + WSEG)                // [128 x 96]
#define WPROJ_OFF (WEMB_OFF + NLAT * DOBS)         // [96 x 128]
#define WIH_OFF   (WPROJ_OFF + DOBS * NLAT)        // [288 x 192]
#define WHH_OFF   (WIH_OFF + 3 * DOBS * 2 * DOBS)  // [288 x 96]

// ---------------------------------------------------------------------------
// f16 WMMA operand helpers (16x16x32).
// A operand: lane L holds row L&15, K-offsets {0..7,16..23} (L<16) or
// {8..15,24..31} (L>=16). B operand symmetric with lanes indexing columns;
// all matmuls are X @ W^T with W stored [n_out, k] row-major, so column n of
// the B operand is row n of W — one loader serves both.
// ---------------------------------------------------------------------------
__device__ __forceinline__ v16h load_ab(const _Float16* base, int ld, int r0,
                                        int k0, int lane) {
  const _Float16* p = base + (r0 + (lane & 15)) * ld + k0 + ((lane & 16) ? 8 : 0);
  v16h v;
#pragma unroll
  for (int i = 0; i < 8; ++i) { v[i] = p[i]; v[8 + i] = p[16 + i]; }
  return v;
}

// C[0:16, n0:n0+16] += A[0:16, 0:K] * W[n0:n0+16, 0:K]^T   (f32 accumulate)
__device__ __forceinline__ v8f mm16(const _Float16* A, int lda,
                                    const _Float16* W, int ldw,
                                    int n0, int K, int lane, v8f acc) {
  for (int k = 0; k < K; k += 32) {
    v16h a = load_ab(A, lda, 0, k, lane);
    v16h b = load_ab(W, ldw, n0, k, lane);
    acc = __builtin_amdgcn_wmma_f32_16x16x32_f16(false, a, false, b,
                                                 (short)0, acc, false, false);
  }
  return acc;
}

// ---------------------------------------------------------------------------
// f32 WMMA operand helpers (16x16x4).  ISA 32-bit A-matrix 16x4 layout:
// lane L holds row L&15; VGPR0 = K {0 | 2}, VGPR1 = K {1 | 3} selected by
// lane>=16.  B (4x16) symmetric with lanes indexing columns.
// ---------------------------------------------------------------------------
__device__ __forceinline__ v2f load_ab_f32(const float* base, int ld, int r0,
                                           int k0, int lane) {
  const float* p = base + (r0 + (lane & 15)) * ld + k0 + ((lane & 16) ? 2 : 0);
  v2f v; v[0] = p[0]; v[1] = p[1];
  return v;
}

__device__ __forceinline__ v8f mmf32(const float* A, int lda,
                                     const float* W, int ldw,
                                     int n0, int K, int lane, v8f acc) {
  for (int k = 0; k < K; k += 4) {
    v2f a = load_ab_f32(A, lda, 0, k, lane);
    v2f b = load_ab_f32(W, ldw, n0, k, lane);
    acc = __builtin_amdgcn_wmma_f32_16x16x4_f32(false, a, false, b,
                                                (short)0, acc, false, false);
  }
  return acc;
}

// ---------------------------------------------------------------------------
// kernel 1: pack weights to f16
// ---------------------------------------------------------------------------
__global__ void pack_kernel(const float* __restrict__ encW1, const float* __restrict__ encW2,
                            const float* __restrict__ decW1, const float* __restrict__ decW2,
                            const float* __restrict__ Wemb,  const float* __restrict__ Wproj,
                            const float* __restrict__ Wih,   const float* __restrict__ Whh,
                            _Float16* __restrict__ ws) {
  int idx = blockIdx.x * blockDim.x + threadIdx.x;
  if (idx < WSEG) {
    ws[ENC1_OFF + idx] = (_Float16)encW1[idx];
    ws[ENC2_OFF + idx] = (_Float16)encW2[idx];
    ws[DEC1_OFF + idx] = (_Float16)decW1[idx];
    ws[DEC2_OFF + idx] = (_Float16)decW2[idx];
  }
  if (idx < NLAT * DOBS)         ws[WEMB_OFF  + idx] = (_Float16)Wemb[idx];
  if (idx < DOBS * NLAT)         ws[WPROJ_OFF + idx] = (_Float16)Wproj[idx];
  if (idx < 3 * DOBS * 2 * DOBS) ws[WIH_OFF   + idx] = (_Float16)Wih[idx];
  if (idx < 3 * DOBS * DOBS)     ws[WHH_OFF   + idx] = (_Float16)Whh[idx];
}

// ---------------------------------------------------------------------------
// kernel 2: expAdt[t] = expm(A * dt[t]), one workgroup (512 thr / 16 waves)
// per timestep.  Full FP32 via v_wmma_f32_16x16x4_f32.  LDS: MF (row-major M),
// PF (row-major P), PT (transposed P) — 192 KB.
// Horner: P = I + (M @ P)/k, k = 12..1.  Squarings: P = P @ P (x6).
// Wave w owns tile-row (w>>1), tile-cols (w&1)*4 .. +4  (4 accumulators/wave
// => no VGPR spills).  Result stored as f16 [n][k] for the scan's f16 WMMA.
// ---------------------------------------------------------------------------
__global__ void expm_kernel(const float* __restrict__ T,
                            const float* __restrict__ A,
                            _Float16* __restrict__ EXP) {
  extern __shared__ unsigned char smem[];
  float* MF = (float*)smem;                 // 128*128 f32
  float* PF = MF + NLAT * NLAT;
  float* PT = PF + NLAT * NLAT;

  const int t    = blockIdx.x;
  const int tid  = threadIdx.x;
  const int lane = tid & 31;
  const int wave = tid >> 5;                // 0..15
  const float dt = (t == 0) ? 0.0f : (T[t] - T[t - 1]);
  const float sc = dt * (1.0f / 64.0f);     // pre-scale by 2^-6 for squaring

  for (int i = tid; i < NLAT * NLAT; i += 512) {
    MF[i] = A[i] * sc;
    int r = i >> 7, c = i & (NLAT - 1);
    float idv = (r == c) ? 1.0f : 0.0f;
    PF[i] = idv; PT[i] = idv;
  }
  __syncthreads();

  const int m0   = (wave >> 1) * 16;        // tile row
  const int tc0  = (wave & 1) * 4;          // first of 4 tile cols
  const int col0 = lane & 15;
  const int rb   = m0 + ((lane & 16) ? 8 : 0);

  // Horner
  for (int k = 12; k >= 1; --k) {
    v8f acc[4];
#pragma unroll
    for (int tc = 0; tc < 4; ++tc) {
      v8f c = {};
      acc[tc] = mmf32(MF + m0 * NLAT, NLAT, PT, NLAT, (tc0 + tc) * 16, NLAT,
                      lane, c);
    }
    __syncthreads();                        // everyone done reading PT
    float inv = 1.0f / (float)k;
#pragma unroll
    for (int tc = 0; tc < 4; ++tc) {
      int cc = (tc0 + tc) * 16 + col0;
#pragma unroll
      for (int i = 0; i < 8; ++i) {
        int rr = rb + i;
        float v = acc[tc][i] * inv + ((rr == cc) ? 1.0f : 0.0f);
        PF[rr * NLAT + cc] = v;
        PT[cc * NLAT + rr] = v;
      }
    }
    __syncthreads();
  }
  // 6 squarings: P = P @ P
  for (int s = 0; s < 6; ++s) {
    v8f acc[4];
#pragma unroll
    for (int tc = 0; tc < 4; ++tc) {
      v8f c = {};
      acc[tc] = mmf32(PF + m0 * NLAT, NLAT, PT, NLAT, (tc0 + tc) * 16, NLAT,
                      lane, c);
    }
    __syncthreads();
#pragma unroll
    for (int tc = 0; tc < 4; ++tc) {
      int cc = (tc0 + tc) * 16 + col0;
#pragma unroll
      for (int i = 0; i < 8; ++i) {
        int rr = rb + i;
        float v = acc[tc][i];
        PF[rr * NLAT + cc] = v;
        PT[cc * NLAT + rr] = v;
      }
    }
    __syncthreads();
  }
  _Float16* dst = EXP + t * (NLAT * NLAT);  // row-major expAdt[n][k], f16
  for (int i = tid; i < NLAT * NLAT; i += 512)
    dst[i] = (_Float16)PF[i];
}

// ---------------------------------------------------------------------------
// resnet: F/H (f32/f16 mirrors, [16][128]) updated in place, scratchH [16][128]
// ---------------------------------------------------------------------------
__device__ __forceinline__ void resnet5(float* F, _Float16* H, _Float16* scratchH,
                                        const _Float16* W1, const _Float16* W2,
                                        const float* B1, const float* B2,
                                        int wave, int lane) {
  const int col0 = lane & 15;
  const int rb   = (lane & 16) ? 8 : 0;
  const int cc   = wave * 16 + col0;
  for (int blk = 0; blk < NB; ++blk) {
    {   // hid = relu(H @ W1^T + b1) -> scratchH
      v8f c = {};
      c = mm16(H, NLAT, W1 + blk * NLAT * NLAT, NLAT, wave * 16, NLAT, lane, c);
      float bb = B1[blk * NLAT + cc];
#pragma unroll
      for (int i = 0; i < 8; ++i) {
        float v = c[i] + bb;
        scratchH[(rb + i) * NLAT + cc] = (_Float16)(v > 0.0f ? v : 0.0f);
      }
    }
    __syncthreads();
    {   // F += scratchH @ W2^T + b2 ; refresh H
      v8f c = {};
      c = mm16(scratchH, NLAT, W2 + blk * NLAT * NLAT, NLAT, wave * 16, NLAT, lane, c);
      float bb = B2[blk * NLAT + cc];
#pragma unroll
      for (int i = 0; i < 8; ++i) {
        float v = F[(rb + i) * NLAT + cc] + c[i] + bb;
        F[(rb + i) * NLAT + cc] = v;
        H[(rb + i) * NLAT + cc] = (_Float16)v;
      }
    }
    __syncthreads();
  }
}

// ---------------------------------------------------------------------------
// kernel 3: persistent sequential scan; block = 16 batch rows, 8 waves.
// ---------------------------------------------------------------------------
__global__ void scan_kernel(const float* __restrict__ X,
                            const unsigned char* __restrict__ mask,
                            const _Float16* __restrict__ ws,
                            const float* __restrict__ b_emb,  const float* __restrict__ b_proj,
                            const float* __restrict__ enc_b1, const float* __restrict__ enc_b2,
                            const float* __restrict__ dec_b1, const float* __restrict__ dec_b2,
                            const float* __restrict__ b_ih,   const float* __restrict__ b_hh,
                            float* __restrict__ out) {
  extern __shared__ unsigned char smem[];
  float*    AF  = (float*)smem;            // [16][128] latent / embed buffer
  float*    BF  = AF + 16 * NLAT;          // [16][128] propagated / decoder
  float*    XHF = BF + 16 * NLAT;          // [16][96]  xhat
  float*    GIF = XHF + 16 * DOBS;         // [16][288] GRU input gates
  float*    GHF = GIF + 16 * 3 * DOBS;     // [16][288] GRU hidden gates
  _Float16* AH  = (_Float16*)(GHF + 16 * 3 * DOBS);
  _Float16* BH  = AH + 16 * NLAT;
  _Float16* XHH = BH + 16 * NLAT;          // [16][96]
  _Float16* UH  = XHH + 16 * DOBS;         // [16][192] GRU input u

  const _Float16* EXP = ws + EXP_OFF;
  const _Float16* EW1 = ws + ENC1_OFF;
  const _Float16* EW2 = ws + ENC2_OFF;
  const _Float16* DW1 = ws + DEC1_OFF;
  const _Float16* DW2 = ws + DEC2_OFF;
  const _Float16* WEM = ws + WEMB_OFF;
  const _Float16* WPJ = ws + WPROJ_OFF;
  const _Float16* WIH = ws + WIH_OFF;
  const _Float16* WHH = ws + WHH_OFF;

  const int tid  = threadIdx.x;
  const int lane = tid & 31;
  const int wave = tid >> 5;
  const int b0   = blockIdx.x * 16;
  const int col0 = lane & 15;
  const int rb   = (lane & 16) ? 8 : 0;

  // z0 = encoder(embed(zeros)) ; embed(0) = b_emb broadcast over rows
  for (int i = tid; i < 16 * NLAT; i += 256) {
    float v = b_emb[i & (NLAT - 1)];
    AF[i] = v; AH[i] = (_Float16)v;
  }
  __syncthreads();
  resnet5(AF, AH, BH, EW1, EW2, enc_b1, enc_b2, wave, lane);

  for (int t = 0; t < LEN; ++t) {
    // prefetch next step's expAdt slab (32 KB; 128 B per thread)
    if (t + 1 < LEN) {
      const _Float16* En = EXP + (t + 1) * (NLAT * NLAT);
      __builtin_prefetch(En + tid * 64, 0, 1);
    }

    // 1) propagate: B = z @ expAdt^T   (expAdt stored [n][k])
    {
      const _Float16* Et = EXP + t * (NLAT * NLAT);
      v8f c = {};
      c = mm16(AH, NLAT, Et, NLAT, wave * 16, NLAT, lane, c);
      int cc = wave * 16 + col0;
#pragma unroll
      for (int i = 0; i < 8; ++i) {
        BF[(rb + i) * NLAT + cc] = c[i];
        BH[(rb + i) * NLAT + cc] = (_Float16)c[i];
      }
    }
    __syncthreads();

    // 2) decoder iResNet (in B, scratch AH)
    resnet5(BF, BH, AH, DW1, DW2, dec_b1, dec_b2, wave, lane);

    // 3) project: xhat_pre = B @ Wproj^T + b_proj   (96 cols -> waves 0..5)
    if (wave < 6) {
      v8f c = {};
      c = mm16(BH, NLAT, WPJ, NLAT, wave * 16, NLAT, lane, c);
      int cc = wave * 16 + col0;
      float bb = b_proj[cc];
#pragma unroll
      for (int i = 0; i < 8; ++i) {
        float v = c[i] + bb;
        XHF[(rb + i) * DOBS + cc] = v;
        XHH[(rb + i) * DOBS + cc] = (_Float16)v;
      }
    }
    __syncthreads();

    // 4) impute + concat mask -> u
    for (int i = tid; i < 16 * DOBS; i += 256) {
      int m = i / DOBS, d = i - m * DOBS;
      size_t g = ((size_t)(b0 + m) * LEN + t) * DOBS + d;
      bool mv = mask[g] != 0;
      float xt = mv ? X[g] : XHF[m * DOBS + d];
      UH[m * 2 * DOBS + d]        = (_Float16)xt;
      UH[m * 2 * DOBS + DOBS + d] = (_Float16)(mv ? 1.0f : 0.0f);
    }
    __syncthreads();

    // 5) GRU gates: gi = u @ W_ih^T + b_ih (K=192), gh = h @ W_hh^T + b_hh (K=96)
    for (int tc = wave; tc < 18; tc += 8) {
      int n0 = tc * 16, cc = n0 + col0;
      v8f ci = {}, ch = {};
      ci = mm16(UH, 2 * DOBS, WIH, 2 * DOBS, n0, 2 * DOBS, lane, ci);
      ch = mm16(XHH, DOBS, WHH, DOBS, n0, DOBS, lane, ch);
      float bi = b_ih[cc], bh = b_hh[cc];
#pragma unroll
      for (int i = 0; i < 8; ++i) {
        GIF[(rb + i) * 3 * DOBS + cc] = ci[i] + bi;
        GHF[(rb + i) * 3 * DOBS + cc] = ch[i] + bh;
      }
    }
    __syncthreads();

    // 6) GRU pointwise + output write
    for (int i = tid; i < 16 * DOBS; i += 256) {
      int m = i / DOBS, d = i - m * DOBS;
      const float* gi = GIF + m * 3 * DOBS;
      const float* gh = GHF + m * 3 * DOBS;
      float r  = 1.0f / (1.0f + expf(-(gi[d] + gh[d])));
      float zg = 1.0f / (1.0f + expf(-(gi[DOBS + d] + gh[DOBS + d])));
      float ng = tanhf(gi[2 * DOBS + d] + r * gh[2 * DOBS + d]);
      float h  = XHF[m * DOBS + d];
      float xo = (1.0f - zg) * ng + zg * h;
      XHF[m * DOBS + d] = xo;
      XHH[m * DOBS + d] = (_Float16)xo;
      out[((size_t)(b0 + m) * LEN + t) * DOBS + d] = xo;
    }
    __syncthreads();

    // 7) embed: A = xhat_post @ W_emb^T + b_emb   (K = 96)
    {
      v8f c = {};
      c = mm16(XHH, DOBS, WEM, DOBS, wave * 16, DOBS, lane, c);
      int cc = wave * 16 + col0;
      float bb = b_emb[cc];
#pragma unroll
      for (int i = 0; i < 8; ++i) {
        float v = c[i] + bb;
        AF[(rb + i) * NLAT + cc] = v;
        AH[(rb + i) * NLAT + cc] = (_Float16)v;
      }
    }
    __syncthreads();

    // 8) encoder iResNet -> next z (in A, scratch BH)
    resnet5(AF, AH, BH, EW1, EW2, enc_b1, enc_b2, wave, lane);
  }
}

// ---------------------------------------------------------------------------
extern "C" void kernel_launch(void* const* d_in, const int* in_sizes, int n_in,
                              void* d_out, int out_size, void* d_ws, size_t ws_size,
                              hipStream_t stream) {
  const float* T      = (const float*)d_in[0];
  const float* X      = (const float*)d_in[1];
  const unsigned char* mask = (const unsigned char*)d_in[2];
  const float* A      = (const float*)d_in[3];
  const float* W_emb  = (const float*)d_in[4];
  const float* b_emb  = (const float*)d_in[5];
  const float* W_proj = (const float*)d_in[6];
  const float* b_proj = (const float*)d_in[7];
  const float* enc_W1 = (const float*)d_in[8];
  const float* enc_b1 = (const float*)d_in[9];
  const float* enc_W2 = (const float*)d_in[10];
  const float* enc_b2 = (const float*)d_in[11];
  const float* dec_W1 = (const float*)d_in[12];
  const float* dec_b1 = (const float*)d_in[13];
  const float* dec_W2 = (const float*)d_in[14];
  const float* dec_b2 = (const float*)d_in[15];
  const float* W_ih   = (const float*)d_in[16];
  const float* W_hh   = (const float*)d_in[17];
  const float* b_ih   = (const float*)d_in[18];
  const float* b_hh   = (const float*)d_in[19];

  _Float16* ws = (_Float16*)d_ws;
  float* out = (float*)d_out;

  // 1) weights -> f16 (largest segment 81920 elems)
  pack_kernel<<<(WSEG + 255) / 256, 256, 0, stream>>>(
      enc_W1, enc_W2, dec_W1, dec_W2, W_emb, W_proj, W_ih, W_hh, ws);

  // 2) per-timestep matrix exponentials, full fp32 WMMA
  //    (512 threads / 16 waves per block, 192 KB LDS)
  expm_kernel<<<LEN, 512, 3 * NLAT * NLAT * (int)sizeof(float), stream>>>(
      T, A, ws + EXP_OFF);

  // 3) sequential scan, 16 batch rows per block (75 KB LDS each)
  size_t scan_lds = (16 * NLAT * 2 + 16 * DOBS + 16 * 3 * DOBS * 2) * sizeof(float) +
                    (16 * NLAT * 2 + 16 * DOBS + 16 * 2 * DOBS) * sizeof(_Float16);
  scan_kernel<<<BATCH / 16, 256, scan_lds, stream>>>(
      X, mask, ws, b_emb, b_proj, enc_b1, enc_b2, dec_b1, dec_b2,
      b_ih, b_hh, out);
}
